// SparseMoe_71141838291440
// MI455X (gfx1250) — matrix operand order repaired and checked by reading, not compiled, not run on hardware
//
#include <hip/hip_runtime.h>
#include <hip/hip_bf16.h>
#include <stddef.h>

// ---------------- problem constants (from reference) ----------------
#define NTOK 4096      // B*S = 2*2048
#define DIM  1024
#define FF   4096
#define NEXP 8
#define EPSL 1e-8f

typedef __attribute__((ext_vector_type(16))) __bf16 bf16x16;
typedef __attribute__((ext_vector_type(8)))  __bf16 bf16x8;
typedef __attribute__((ext_vector_type(8)))  float  f32x8;

// ---------------- 0) zero output + counters ----------------
__global__ void moe_zero(float* __restrict__ out, int* __restrict__ cnt, int n) {
    int i = blockIdx.x * blockDim.x + threadIdx.x;
    if (i < n) out[i] = 0.0f;
    if (i < NEXP) cnt[i] = 0;
}

// ---------------- 1) weight fp32 -> bf16 with transpose ----------------
// w1 [E][D][F] -> w1t [E][F][D]   (so GEMM1 B-frag K(=d) is contiguous)
__global__ void conv_w1(const float* __restrict__ w1, __bf16* __restrict__ w1t) {
    size_t i = (size_t)blockIdx.x * blockDim.x + threadIdx.x;   // over E*FF*DIM, d fastest
    int d = (int)(i & (DIM - 1));
    size_t ef = i >> 10;
    int f = (int)(ef & (FF - 1));
    int e = (int)(ef >> 12);
    w1t[i] = (__bf16)w1[(((size_t)e * DIM + d) * FF) + f];
}
// w2 [E][F][D] -> w2t [E][D][F]   (so GEMM2 B-frag K(=f) is contiguous)
__global__ void conv_w2(const float* __restrict__ w2, __bf16* __restrict__ w2t) {
    size_t i = (size_t)blockIdx.x * blockDim.x + threadIdx.x;   // over E*DIM*FF, f fastest
    int f = (int)(i & (FF - 1));
    size_t ed = i >> 12;
    int d = (int)(ed & (DIM - 1));
    int e = (int)(ed >> 10);
    w2t[i] = (__bf16)w2[(((size_t)e * FF + f) * DIM) + d];
}

// ---------------- 2) router: logits, top-2, softmax, gather ----------------
__global__ __launch_bounds__(256) void moe_router(
    const float* __restrict__ x, const float* __restrict__ noise,
    const float* __restrict__ gate_w, const float* __restrict__ gate_b,
    const float* __restrict__ noise_w, const float* __restrict__ noise_b,
    float* __restrict__ scores, int* __restrict__ cnt,
    int* __restrict__ tok_list, float* __restrict__ tok_score)
{
    const int wv   = threadIdx.x >> 5;          // wave32
    const int lane = threadIdx.x & 31;
    const int t    = blockIdx.x * 8 + wv;

    float accg[NEXP], accn[NEXP];
#pragma unroll
    for (int e = 0; e < NEXP; ++e) { accg[e] = 0.f; accn[e] = 0.f; }

    const float* xr = x + (size_t)t * DIM;
    for (int d = lane; d < DIM; d += 32) {
        float xv = xr[d];
#pragma unroll
        for (int e = 0; e < NEXP; ++e) {
            accg[e] += xv * gate_w[d * NEXP + e];
            accn[e] += xv * noise_w[d * NEXP + e];
        }
    }
    float logit[NEXP];
#pragma unroll
    for (int e = 0; e < NEXP; ++e) {
        float v = accg[e] + accn[e];
#pragma unroll
        for (int off = 16; off > 0; off >>= 1) v += __shfl_xor(v, off, 32);
        logit[e] = v + gate_b[e] + noise_b[e] + noise[(size_t)t * NEXP + e];
    }
    if (lane == 0) {
        int i0 = 0;
#pragma unroll
        for (int e = 1; e < NEXP; ++e) if (logit[e] > logit[i0]) i0 = e;   // ties -> lowest idx
        int i1 = -1;
#pragma unroll
        for (int e = 0; e < NEXP; ++e) {
            if (e == i0) continue;
            if (i1 < 0 || logit[e] > logit[i1]) i1 = e;
        }
        float m  = logit[i0];
        float e0 = __expf(logit[i0] - m), e1 = __expf(logit[i1] - m);
        float inv = 1.0f / (e0 + e1);
        float s0 = e0 * inv, s1 = e1 * inv;
#pragma unroll
        for (int e = 0; e < NEXP; ++e) scores[(size_t)t * NEXP + e] = 0.0f;
        scores[(size_t)t * NEXP + i0] = s0;
        scores[(size_t)t * NEXP + i1] = s1;
        int p0 = atomicAdd(&cnt[i0], 1);
        tok_list[i0 * NTOK + p0] = t;  tok_score[i0 * NTOK + p0] = s0;
        int p1 = atomicAdd(&cnt[i1], 1);
        tok_list[i1 * NTOK + p1] = t;  tok_score[i1 * NTOK + p1] = s1;
    }
}

// ---------------- 3) aux loss: deterministic fixed-order reduction ----------------
__global__ __launch_bounds__(256) void moe_aux(const float* __restrict__ scores,
                                               float* __restrict__ aux_out)
{
    __shared__ float red[256];
    __shared__ float imp[NEXP];
    const int tid = threadIdx.x;
    for (int e = 0; e < NEXP; ++e) {
        float s = 0.f;
        for (int t = tid; t < NTOK; t += 256) s += scores[(size_t)t * NEXP + e];
        red[tid] = s;  __syncthreads();
        for (int w = 128; w > 0; w >>= 1) {
            if (tid < w) red[tid] += red[tid + w];
            __syncthreads();
        }
        if (tid == 0) imp[e] = red[0] / (float)NTOK;
        __syncthreads();
    }
    if (tid == 0) {
        const float u = 1.0f / NEXP, lu = __logf(1.0f / NEXP);
        float loss = 0.f;
        for (int e = 0; e < NEXP; ++e) loss += u * (lu - __logf(imp[e] + EPSL));
        aux_out[0] = loss;
    }
}

// ---------------- 4) fused expert FFN: silu(x@w1+b1)@w2+b2, scaled scatter ----------------
// grid: (NTOK/16 tiles, NEXP experts), block: 256 (8 waves of 32).
// M-tile = 16 gathered tokens. Wave w owns output cols [w*128, w*128+128).
__global__ __launch_bounds__(256) void moe_ffn(
    const __bf16* __restrict__ w1t,        // [E][F][D]
    const __bf16* __restrict__ w2t,        // [E][D][F]
    const float*  __restrict__ b1,         // [E][F]
    const float*  __restrict__ b2,         // [E][D]
    const float*  __restrict__ x,          // [T][D]
    const int*    __restrict__ cnt,        // [E]
    const int*    __restrict__ tok_list,   // [E][T]
    const float*  __restrict__ tok_score,  // [E][T]
    float*        __restrict__ out)        // [T][D]
{
    const int e      = blockIdx.y;
    const int tile   = blockIdx.x;
    const int nvalid = cnt[e] - tile * 16;
    if (nvalid <= 0) return;                       // uniform exit: EXEC full below

    const int tid  = threadIdx.x;
    const int wv   = tid >> 5;
    const int lane = tid & 31;
    const int nlo  = lane & 15;                    // N (or M-row) index within fragment
    const int hi   = lane >> 4;                    // K-half / M-half selector

    __shared__ __align__(16) __bf16 xs[16 * DIM];  // 32 KB: x tile (bf16, row-major)
    __shared__ __align__(16) __bf16 hs[16 * 128];  //  4 KB: h chunk (bf16, row-major)
    __shared__ int   toks[16];
    __shared__ float gsc[16];

    if (tid < 16) {
        bool v = tid < nvalid;
        toks[tid] = v ? tok_list[e * NTOK + tile * 16 + tid] : 0;
        gsc[tid]  = v ? tok_score[e * NTOK + tile * 16 + tid] : 0.0f;
    }
    __syncthreads();
    for (int i = tid; i < 16 * DIM; i += 256) {    // coalesced gather + cvt to bf16
        int m = i >> 10, d = i & (DIM - 1);
        float v = (m < nvalid) ? x[(size_t)toks[m] * DIM + d] : 0.0f;
        xs[i] = (__bf16)v;
    }
    __syncthreads();

    f32x8 accD[8];
#pragma unroll
    for (int i = 0; i < 8; ++i) accD[i] = (f32x8){0.f,0.f,0.f,0.f,0.f,0.f,0.f,0.f};

    // A-fragment base for GEMM1: lane row = nlo, 16 contiguous K at hi*16
    const __bf16* aX = xs + nlo * DIM + hi * 16;

    for (int fc = 0; fc < FF / 128; ++fc) {
        // ---- Phase A: wave wv computes h tile, F-cols [fc*128 + wv*16, +16) ----
        const int fcol = fc * 128 + wv * 16 + nlo;                  // this lane's B column
        const __bf16* bW1 = w1t + ((size_t)e * FF + fcol) * DIM + hi * 16;

        // prefetch next chunk's weight streams into cache (global_prefetch_b8)
        if (fc + 1 < FF / 128) {
            __builtin_prefetch(bW1 + (size_t)128 * DIM, 0, 3);               // next w1 column
            __builtin_prefetch(w2t + ((size_t)e * DIM + wv * 128 + nlo) * FF
                                   + (fc + 1) * 128 + hi * 16, 0, 3);        // next w2 row
        }

        f32x8 acch = (f32x8){0.f,0.f,0.f,0.f,0.f,0.f,0.f,0.f};
#pragma unroll 8
        for (int kb = 0; kb < DIM; kb += 32) {
            bf16x16 a;
            {
                bf16x8 a0 = *(const bf16x8*)(aX + kb);
                bf16x8 a1 = *(const bf16x8*)(aX + kb + 8);
#pragma unroll
                for (int j = 0; j < 8; ++j) { a[j] = a0[j]; a[8 + j] = a1[j]; }
            }
            bf16x16 b = *(const bf16x16*)(bW1 + kb);               // 16 contiguous K
            acch = __builtin_amdgcn_wmma_f32_16x16x32_bf16(
                       false, a, false, b, (short)0, acch, false, false);
        }
        // bias + SiLU, write h to LDS (C-layout: VGPR r, M = hi*8 + r, N = nlo)
        // SiLU via v_rcp_f32 (bf16-quantized downstream; precise divide is wasted VALU)
        const float bb1 = b1[e * FF + fcol];
#pragma unroll
        for (int r = 0; r < 8; ++r) {
            int m = hi * 8 + r;
            float v = acch[r] + bb1;
            float s = v * __builtin_amdgcn_rcpf(1.0f + __expf(-v));
            hs[m * 128 + wv * 16 + nlo] = (__bf16)s;
        }
        __syncthreads();

        // ---- Phase B: out[., wv*128 .. +128) += h(16x128) @ w2 chunk ----
        const __bf16* aH = hs + nlo * 128 + hi * 16;
#pragma unroll
        for (int nt = 0; nt < 8; ++nt) {
            const int dcol = wv * 128 + nt * 16 + nlo;
            const __bf16* bW2 = w2t + ((size_t)e * DIM + dcol) * FF + fc * 128 + hi * 16;
#pragma unroll
            for (int kk = 0; kk < 128; kk += 32) {
                bf16x16 a;
                {
                    bf16x8 a0 = *(const bf16x8*)(aH + kk);
                    bf16x8 a1 = *(const bf16x8*)(aH + kk + 8);
#pragma unroll
                    for (int j = 0; j < 8; ++j) { a[j] = a0[j]; a[8 + j] = a1[j]; }
                }
                bf16x16 b = *(const bf16x16*)(bW2 + kk);
                accD[nt] = __builtin_amdgcn_wmma_f32_16x16x32_bf16(
                               false, a, false, b, (short)0, accD[nt], false, false);
            }
        }
        __syncthreads();   // hs reused next iteration
    }

    // ---- epilogue: out[t] += score * (h@w2 + b2); padded rows have score 0 ----
#pragma unroll
    for (int nt = 0; nt < 8; ++nt) {
        const int dcol = wv * 128 + nt * 16 + nlo;
        const float bb2 = b2[e * DIM + dcol];
#pragma unroll
        for (int r = 0; r < 8; ++r) {
            int m = hi * 8 + r;
            float v = (accD[nt][r] + bb2) * gsc[m];
            atomicAdd(&out[(size_t)toks[m] * DIM + dcol], v);   // 2 commutative adds/token
        }
    }
}

// ---------------- launcher ----------------
extern "C" void kernel_launch(void* const* d_in, const int* in_sizes, int n_in,
                              void* d_out, int out_size, void* d_ws, size_t ws_size,
                              hipStream_t stream) {
    const float* x       = (const float*)d_in[0];
    const float* noise   = (const float*)d_in[1];
    const float* gate_w  = (const float*)d_in[2];
    const float* gate_b  = (const float*)d_in[3];
    const float* noise_w = (const float*)d_in[4];
    const float* noise_b = (const float*)d_in[5];
    const float* w1      = (const float*)d_in[6];
    const float* b1      = (const float*)d_in[7];
    const float* w2      = (const float*)d_in[8];
    const float* b2      = (const float*)d_in[9];
    float* out = (float*)d_out;

    // workspace layout (assumes ws_size >= ~129 MB)
    char* ws = (char*)d_ws;
    const size_t W1T_BYTES = (size_t)NEXP * FF * DIM * 2;   // 64 MiB
    const size_t W2T_BYTES = (size_t)NEXP * DIM * FF * 2;   // 64 MiB
    __bf16* w1t      = (__bf16*)(ws);
    __bf16* w2t      = (__bf16*)(ws + W1T_BYTES);
    float*  scores   = (float*)(ws + W1T_BYTES + W2T_BYTES);                 // T*E f32
    int*    cnt      = (int*)  (ws + W1T_BYTES + W2T_BYTES + 131072);
    int*    tok_list = (int*)  (ws + W1T_BYTES + W2T_BYTES + 131072 + 256);  // E*T i32
    float*  tok_score= (float*)(ws + W1T_BYTES + W2T_BYTES + 131072 + 256 + 131072);

    const int nOut = NTOK * DIM;
    moe_zero<<<(nOut + 255) / 256, 256, 0, stream>>>(out, cnt, nOut);
    conv_w1<<<(int)((size_t)NEXP * FF * DIM / 256), 256, 0, stream>>>(w1, w1t);
    conv_w2<<<(int)((size_t)NEXP * DIM * FF / 256), 256, 0, stream>>>(w2, w2t);
    moe_router<<<NTOK / 8, 256, 0, stream>>>(x, noise, gate_w, gate_b, noise_w, noise_b,
                                             scores, cnt, tok_list, tok_score);
    moe_aux<<<1, 256, 0, stream>>>(scores, out + nOut);
    moe_ffn<<<dim3(NTOK / 16, NEXP), 256, 0, stream>>>(w1t, w2t, b1, b2, x,
                                                       cnt, tok_list, tok_score, out);
}
